// GPT_61392262529458
// MI455X (gfx1250) — matrix operand order repaired and checked by reading, not compiled, not run on hardware
//
#include <hip/hip_runtime.h>
#include <hip/hip_bf16.h>
#include <math.h>

// ---------------- model constants ----------------
constexpr int Vv  = 50257;
constexpr int E   = 1024;
constexpr int Hh  = 16;
constexpr int Ll  = 8;
constexpr int Bb  = 2;
constexpr int Tt  = 2048;
constexpr int HD  = 64;      // E / H
constexpr int FF  = 4096;    // 4*E
constexpr int BT  = Bb * Tt; // 4096 rows

typedef _Float16 half16 __attribute__((ext_vector_type(16)));
typedef _Float16 half8  __attribute__((ext_vector_type(8)));
typedef _Float16 half4  __attribute__((ext_vector_type(4)));
typedef float    float8 __attribute__((ext_vector_type(8)));

// ---------------------------------------------------------------------------
// WMMA fragment loaders (CDNA5 16x16x32 f16; contiguous k order used by this file for both A and B)
//
// A (16x32, MxK): lanes 0-15 -> row M=lane; e0..15 = K(koff + e),
//                 koff = (lane<16 ? 0 : 16).
// B (32x16, KxN): col N = (lane&15); e0..15 = K(koff2 + e), koff2 = lane<16?0:16.
// C/D (16x16 f32): VGPR r, lanes 0-15 -> M=r, N=lane; lanes 16-31 -> M=r+8.
//
// LDS layouts make every fragment two 16B ds_load_b128:
//   A-tile row-major  [m][k]  (row stride 40 halves = 80B, 16B-multiple)
//   B-tile TRANSPOSED [n][k]
// ---------------------------------------------------------------------------
__device__ __forceinline__ half16 load_a_frag(const _Float16* lds, int stride, int lane) {
  const int m    = lane & 15;
  const int koff = (lane < 16) ? 0 : 16;
  const _Float16* rowp = lds + m * stride;
  half8 lo = *(const half8*)(rowp + koff);
  half8 hi = *(const half8*)(rowp + koff + 8);
  return __builtin_shufflevector(lo, hi, 0,1,2,3,4,5,6,7,8,9,10,11,12,13,14,15);
}

__device__ __forceinline__ half16 load_bT_frag(const _Float16* ldsT, int stride, int lane, int nbase) {
  const int n    = nbase + (lane & 15);
  const int koff = (lane < 16) ? 0 : 16;
  const _Float16* p = ldsT + n * stride + koff;
  half8 lo = *(const half8*)(p);
  half8 hi = *(const half8*)(p + 8);
  return __builtin_shufflevector(lo, hi, 0,1,2,3,4,5,6,7,8,9,10,11,12,13,14,15);
}

__device__ __forceinline__ float8 wmma_f16(half16 a, half16 b, float8 c) {
  return __builtin_amdgcn_wmma_f32_16x16x32_f16(false, a, false, b, (short)0, c, false, false);
}

__device__ __forceinline__ float gelu_exact(float v) {
  return 0.5f * v * (1.0f + erff(v * 0.70710678118654752f));
}

__device__ __forceinline__ half8 cvt8(const float4 a, const float4 b) {
  half8 h;
  h[0] = (_Float16)a.x; h[1] = (_Float16)a.y; h[2] = (_Float16)a.z; h[3] = (_Float16)a.w;
  h[4] = (_Float16)b.x; h[5] = (_Float16)b.y; h[6] = (_Float16)b.z; h[7] = (_Float16)b.w;
  return h;
}

// ---------------------------------------------------------------------------
// Embedding: x[b,t,:] = wte[idx[b,t],:] + wpe[t,:]
// ---------------------------------------------------------------------------
__global__ __launch_bounds__(256) void embed_kernel(const int* __restrict__ idx,
                                                    const float* __restrict__ wte,
                                                    const float* __restrict__ wpe,
                                                    float* __restrict__ x) {
  const int row = blockIdx.x;            // 0..BT-1
  const int t   = row & (Tt - 1);
  const int tok = idx[row];
  const float4* we = (const float4*)(wte + (size_t)tok * E);
  const float4* pe = (const float4*)(wpe + (size_t)t * E);
  float4* xo = (float4*)(x + (size_t)row * E);
  float4 a = we[threadIdx.x], b = pe[threadIdx.x];
  xo[threadIdx.x] = make_float4(a.x + b.x, a.y + b.y, a.z + b.z, a.w + b.w);
}

// ---------------------------------------------------------------------------
// LayerNorm over E=1024 (one block per row, float4 per thread)
// ---------------------------------------------------------------------------
__global__ __launch_bounds__(256) void ln_kernel(const float* __restrict__ x,
                                                 const float* __restrict__ g,
                                                 const float* __restrict__ b,
                                                 float* __restrict__ out) {
  const int row = blockIdx.x;
  const int tid = threadIdx.x;
  __shared__ float red[256];

  float4 v = ((const float4*)(x + (size_t)row * E))[tid];
  float s = v.x + v.y + v.z + v.w;
  red[tid] = s; __syncthreads();
  for (int st = 128; st > 0; st >>= 1) { if (tid < st) red[tid] += red[tid + st]; __syncthreads(); }
  const float mean = red[0] * (1.0f / E);
  __syncthreads();

  float dx = v.x - mean, dy = v.y - mean, dz = v.z - mean, dw = v.w - mean;
  red[tid] = dx * dx + dy * dy + dz * dz + dw * dw; __syncthreads();
  for (int st = 128; st > 0; st >>= 1) { if (tid < st) red[tid] += red[tid + st]; __syncthreads(); }
  const float inv = 1.0f / sqrtf(red[0] * (1.0f / E) + 1e-5f);

  float4 gg = ((const float4*)g)[tid];
  float4 bb = ((const float4*)b)[tid];
  float4 o = make_float4(gg.x * (dx * inv) + bb.x, gg.y * (dy * inv) + bb.y,
                         gg.z * (dz * inv) + bb.z, gg.w * (dw * inv) + bb.w);
  ((float4*)(out + (size_t)row * E))[tid] = o;
}

// ---------------------------------------------------------------------------
// WMMA GEMM: C[M,N] = A[M,K] @ B[K,N] (+bias)(+resid)(gelu). fp32 in/out,
// f16 operands in LDS, fp32 accumulate.
// 128 thr (4 waves), 64(M) x 128(N) tile, K-step 32, double-buffered LDS
// (one barrier per K-step; next tile's global loads overlap WMMAs).
// Each wave: 32x64 -> 8 accumulators, 8 WMMAs per step.
// transB: B stored [N,K] (lm_head; ragged N guarded per row).
// Non-trans call sites guarantee N % 128 == 0.
// ---------------------------------------------------------------------------
__global__ __launch_bounds__(128) void gemm_wmma(const float* __restrict__ A,
                                                 const float* __restrict__ Bm,
                                                 const float* __restrict__ bias,
                                                 const float* __restrict__ resid,
                                                 float* __restrict__ C,
                                                 int M, int N, int K,
                                                 int transB, int doGelu) {
  __shared__ alignas(16) _Float16 As [2][64 * 40];    // [m][k]
  __shared__ alignas(16) _Float16 BsT[2][128 * 40];   // [n][k] (transposed)

  const int tid   = threadIdx.x;
  const int lane  = tid & 31;
  const int wave  = tid >> 5;
  const int tileM = blockIdx.y * 64;
  const int tileN = blockIdx.x * 128;
  const int wrow  = (wave >> 1) * 32;    // 0 or 32
  const int wcol  = (wave & 1) * 64;     // 0 or 64

  float8 acc[2][4];
#pragma unroll
  for (int i = 0; i < 2; ++i)
#pragma unroll
    for (int j = 0; j < 4; ++j)
#pragma unroll
      for (int r = 0; r < 8; ++r) acc[i][j][r] = 0.f;

  auto stage = [&](int k0, int buf) {
    // A tile 64x32: thread -> 16 contiguous k of one row
    {
      int r = tid >> 1, cb = (tid & 1) * 16;
      const float4* ap = (const float4*)(A + (size_t)(tileM + r) * K + k0 + cb);
      *(half8*)(&As[buf][r * 40 + cb])     = cvt8(ap[0], ap[1]);
      *(half8*)(&As[buf][r * 40 + cb + 8]) = cvt8(ap[2], ap[3]);
    }
    // B tile 32x128 -> BsT[n][k]
    if (transB) {
      int n = tid;                        // one [N,K] row per thread
      int gn = tileN + n;
      _Float16* dst = &BsT[buf][n * 40];
      if (gn < N) {
        const float4* bp = (const float4*)(Bm + (size_t)gn * K + k0);
        *(half8*)(dst)      = cvt8(bp[0], bp[1]);
        *(half8*)(dst + 8)  = cvt8(bp[2], bp[3]);
        *(half8*)(dst + 16) = cvt8(bp[4], bp[5]);
        *(half8*)(dst + 24) = cvt8(bp[6], bp[7]);
      } else {
        half8 z;
#pragma unroll
        for (int e = 0; e < 8; ++e) z[e] = (_Float16)0.f;
        *(half8*)(dst) = z; *(half8*)(dst + 8) = z;
        *(half8*)(dst + 16) = z; *(half8*)(dst + 24) = z;
      }
    } else {
      // 4k x 8n register transpose per thread (N % 128 == 0 here)
      int kb = (tid & 7) * 4, nb = (tid >> 3) * 8;
      float4 f[4][2];
#pragma unroll
      for (int j = 0; j < 4; ++j) {
        const float4* bp = (const float4*)(Bm + (size_t)(k0 + kb + j) * N + tileN + nb);
        f[j][0] = bp[0];
        f[j][1] = bp[1];
      }
#pragma unroll
      for (int i = 0; i < 8; ++i) {
        int half_i = i >> 2, sub = i & 3;
        half4 hv;
#pragma unroll
        for (int j = 0; j < 4; ++j) {
          float fv = sub == 0 ? f[j][half_i].x : sub == 1 ? f[j][half_i].y
                   : sub == 2 ? f[j][half_i].z : f[j][half_i].w;
          hv[j] = (_Float16)fv;
        }
        *(half4*)(&BsT[buf][(nb + i) * 40 + kb]) = hv;
      }
    }
  };

  stage(0, 0);
  __syncthreads();

  int buf = 0;
  for (int k0 = 0; k0 < K; k0 += 32, buf ^= 1) {
    half16 a0 = load_a_frag(&As[buf][wrow * 40],        40, lane);
    half16 a1 = load_a_frag(&As[buf][(wrow + 16) * 40], 40, lane);
    half16 b0 = load_bT_frag(&BsT[buf][0], 40, lane, wcol);
    half16 b1 = load_bT_frag(&BsT[buf][0], 40, lane, wcol + 16);
    half16 b2 = load_bT_frag(&BsT[buf][0], 40, lane, wcol + 32);
    half16 b3 = load_bT_frag(&BsT[buf][0], 40, lane, wcol + 48);

    if (k0 + 32 < K) stage(k0 + 32, buf ^ 1);   // overlaps with WMMAs below

    acc[0][0] = wmma_f16(a0, b0, acc[0][0]);
    acc[0][1] = wmma_f16(a0, b1, acc[0][1]);
    acc[0][2] = wmma_f16(a0, b2, acc[0][2]);
    acc[0][3] = wmma_f16(a0, b3, acc[0][3]);
    acc[1][0] = wmma_f16(a1, b0, acc[1][0]);
    acc[1][1] = wmma_f16(a1, b1, acc[1][1]);
    acc[1][2] = wmma_f16(a1, b2, acc[1][2]);
    acc[1][3] = wmma_f16(a1, b3, acc[1][3]);
    __syncthreads();
  }

  // epilogue (C layout: VGPR r -> row r (+8 for lanes>=16), col = lane&15)
  const int rlo = (lane < 16) ? 0 : 8;
  const int nc  = lane & 15;
#pragma unroll
  for (int fi = 0; fi < 2; ++fi) {
#pragma unroll
    for (int fj = 0; fj < 4; ++fj) {
      int col = tileN + wcol + fj * 16 + nc;
      if (col >= N) continue;
#pragma unroll
      for (int r = 0; r < 8; ++r) {
        int row = tileM + wrow + fi * 16 + rlo + r;
        float v = acc[fi][fj][r];
        if (bias)  v += bias[col];
        if (resid) v += resid[(size_t)row * N + col];
        if (doGelu) v = gelu_exact(v);
        C[(size_t)row * N + col] = v;
      }
    }
  }
}

// ---------------------------------------------------------------------------
// Causal flash attention, one wave per (b,h, 16-query tile). head_dim=64.
// S = (Q/8) K^T via WMMA, online softmax fp32, O += P V via WMMA.
// ---------------------------------------------------------------------------
__global__ __launch_bounds__(32) void attn_kernel(const float* __restrict__ qkv,
                                                  float* __restrict__ y) {
  const int lane = threadIdx.x;
  const int qt   = blockIdx.x;        // query tile (16 rows)
  const int bh   = blockIdx.y;
  const int b    = bh / Hh;
  const int h    = bh % Hh;

  __shared__ alignas(16) _Float16 Qs [16][72];   // [q][d]
  __shared__ alignas(16) _Float16 Ks [32][72];   // [key][d]
  __shared__ alignas(16) _Float16 VsT[64][40];   // [d][key]  (transposed)
  __shared__ alignas(16) _Float16 Ps [16][40];   // [q][key]
  __shared__ float Ss[16][36];
  __shared__ float mrow[16], lrow[16], arow[16];

  // load Q (scaled by 1/sqrt(64)=0.125, exact): 2 lanes per row
  {
    int r  = lane >> 1;
    int c0 = (lane & 1) * 32;
    int t  = qt * 16 + r;
    const float4* qp = (const float4*)(qkv + (size_t)(b * Tt + t) * (3 * E) + h * HD + c0);
#pragma unroll
    for (int ch = 0; ch < 4; ++ch) {
      float4 f0 = qp[2 * ch], f1 = qp[2 * ch + 1];
      f0.x *= 0.125f; f0.y *= 0.125f; f0.z *= 0.125f; f0.w *= 0.125f;
      f1.x *= 0.125f; f1.y *= 0.125f; f1.z *= 0.125f; f1.w *= 0.125f;
      *(half8*)(&Qs[r][c0 + ch * 8]) = cvt8(f0, f1);
    }
  }
  if (lane < 16) { mrow[lane] = -1e30f; lrow[lane] = 0.f; }

  float8 o[4];
#pragma unroll
  for (int t4 = 0; t4 < 4; ++t4)
#pragma unroll
    for (int r = 0; r < 8; ++r) o[t4][r] = 0.f;
  __syncthreads();

  const int rlo = (lane < 16) ? 0 : 8;
  const int nc  = lane & 15;
  const int ntiles = (qt >> 1) + 1;   // 32-key tiles covering keys <= qt*16+15

  for (int kt = 0; kt < ntiles; ++kt) {
    // stage K row-major: one key row per lane
    {
      int t = kt * 32 + lane;
      const float4* kp = (const float4*)(qkv + (size_t)(b * Tt + t) * (3 * E) + E + h * HD);
#pragma unroll
      for (int ch = 0; ch < 8; ++ch)
        *(half8*)(&Ks[lane][ch * 8]) = cvt8(kp[2 * ch], kp[2 * ch + 1]);
    }
    // stage V transposed: lane owns d-columns {lane, lane+32}
    {
      const float* vb = qkv + (size_t)(b * Tt + kt * 32) * (3 * E) + 2 * E + h * HD;
#pragma unroll
      for (int dsel = 0; dsel < 2; ++dsel) {
        int d = lane + dsel * 32;
#pragma unroll
        for (int ch = 0; ch < 4; ++ch) {
          half8 hv;
#pragma unroll
          for (int j = 0; j < 8; ++j)
            hv[j] = (_Float16)vb[(size_t)(ch * 8 + j) * (3 * E) + d];
          *(half8*)(&VsT[d][ch * 8]) = hv;
        }
      }
    }
    __syncthreads();

    // S = Q K^T (16x32), masked, into LDS
#pragma unroll
    for (int nh = 0; nh < 2; ++nh) {
      float8 s;
#pragma unroll
      for (int r = 0; r < 8; ++r) s[r] = 0.f;
#pragma unroll
      for (int kk = 0; kk < 64; kk += 32) {
        half16 a = load_a_frag(&Qs[0][0] + kk, 72, lane);
        int n     = nh * 16 + nc;
        int koff2 = (lane < 16) ? 0 : 16;
        const _Float16* p = &Ks[n][kk + koff2];
        half8 blo = *(const half8*)(p);
        half8 bhi = *(const half8*)(p + 8);
        half16 bf = __builtin_shufflevector(blo, bhi, 0,1,2,3,4,5,6,7,8,9,10,11,12,13,14,15);
        s = wmma_f16(a, bf, s);
      }
      int col   = nh * 16 + nc;
      int kglob = kt * 32 + col;
#pragma unroll
      for (int r = 0; r < 8; ++r) {
        int qglob = qt * 16 + rlo + r;
        Ss[rlo + r][col] = (kglob <= qglob) ? s[r] : -1e30f;
      }
    }
    __syncthreads();

    // online softmax (lanes 0-15, one row each); P written as half8
    if (lane < 16) {
      int r = lane;
      float mx = mrow[r];
#pragma unroll
      for (int c = 0; c < 32; ++c) mx = fmaxf(mx, Ss[r][c]);
      float al = expf(mrow[r] - mx);
      float ps = 0.f;
#pragma unroll
      for (int ch = 0; ch < 4; ++ch) {
        half8 hv;
#pragma unroll
        for (int j = 0; j < 8; ++j) {
          float p = expf(Ss[r][ch * 8 + j] - mx);
          ps += p;
          hv[j] = (_Float16)p;
        }
        *(half8*)(&Ps[r][ch * 8]) = hv;
      }
      lrow[r] = lrow[r] * al + ps;
      mrow[r] = mx;
      arow[r] = al;
    }
    __syncthreads();

    // O = O*alpha + P V
    {
      half16 a = load_a_frag(&Ps[0][0], 40, lane);
#pragma unroll
      for (int t4 = 0; t4 < 4; ++t4) {
        half16 bf = load_bT_frag(&VsT[0][0], 40, lane, t4 * 16);
#pragma unroll
        for (int r = 0; r < 8; ++r) o[t4][r] *= arow[rlo + r];
        o[t4] = wmma_f16(a, bf, o[t4]);
      }
    }
    __syncthreads();
  }

  // normalize and write y[b, q, h*64 + d]
#pragma unroll
  for (int t4 = 0; t4 < 4; ++t4)
#pragma unroll
    for (int r = 0; r < 8; ++r) {
      int qglob = qt * 16 + rlo + r;
      float val = o[t4][r] / lrow[rlo + r];
      y[(size_t)(b * Tt + qglob) * E + h * HD + t4 * 16 + nc] = val;
    }
}

// ---------------------------------------------------------------------------
// Per-row NLL: nll[row] = logsumexp(logits[row,:]) - logits[row, target]
// ---------------------------------------------------------------------------
__global__ __launch_bounds__(256) void nll_kernel(const float* __restrict__ logits,
                                                  const int* __restrict__ targets,
                                                  float* __restrict__ nll) {
  const int row = blockIdx.x;
  const int tid = threadIdx.x;
  const float* lr = logits + (size_t)row * Vv;
  __shared__ float red[256];

  float mx = -1e30f;
  for (int c = tid; c < Vv; c += 256) mx = fmaxf(mx, lr[c]);
  red[tid] = mx; __syncthreads();
  for (int st = 128; st > 0; st >>= 1) { if (tid < st) red[tid] = fmaxf(red[tid], red[tid + st]); __syncthreads(); }
  const float m = red[0];
  __syncthreads();

  float s = 0.f;
  for (int c = tid; c < Vv; c += 256) s += expf(lr[c] - m);
  red[tid] = s; __syncthreads();
  for (int st = 128; st > 0; st >>= 1) { if (tid < st) red[tid] += red[tid + st]; __syncthreads(); }

  if (tid == 0) {
    int tgt = targets[row];
    nll[row] = m + logf(red[0]) - lr[tgt];
  }
}

__global__ __launch_bounds__(256) void loss_reduce(const float* __restrict__ nll,
                                                   float* __restrict__ lossOut) {
  const int tid = threadIdx.x;
  __shared__ float red[256];
  float s = 0.f;
  for (int i = tid; i < BT; i += 256) s += nll[i];
  red[tid] = s; __syncthreads();
  for (int st = 128; st > 0; st >>= 1) { if (tid < st) red[tid] += red[tid + st]; __syncthreads(); }
  if (tid == 0) lossOut[0] = red[0] * (1.0f / BT);
}

// ---------------------------------------------------------------------------
extern "C" void kernel_launch(void* const* d_in, const int* in_sizes, int n_in,
                              void* d_out, int out_size, void* d_ws, size_t ws_size,
                              hipStream_t stream) {
  const int*   idx     = (const int*)  d_in[0];
  const int*   targets = (const int*)  d_in[1];
  const float* wte     = (const float*)d_in[2];
  const float* wpe     = (const float*)d_in[3];
  const float* qkv_w   = (const float*)d_in[4];
  const float* proj_w  = (const float*)d_in[5];
  const float* proj_b  = (const float*)d_in[6];
  const float* ln1_g   = (const float*)d_in[7];
  const float* ln1_b   = (const float*)d_in[8];
  const float* ln2_g   = (const float*)d_in[9];
  const float* ln2_b   = (const float*)d_in[10];
  const float* fc1_w   = (const float*)d_in[11];
  const float* fc1_b   = (const float*)d_in[12];
  const float* fc2_w   = (const float*)d_in[13];
  const float* fc2_b   = (const float*)d_in[14];
  const float* lnf_g   = (const float*)d_in[15];
  const float* lnf_b   = (const float*)d_in[16];

  float* ws   = (float*)d_ws;
  float* x    = ws;                           // [BT, E]
  float* h    = x    + (size_t)BT * E;        // [BT, E]
  float* qkv  = h    + (size_t)BT * E;        // [BT, 3E]
  float* yb   = qkv  + (size_t)BT * 3 * E;    // [BT, E]
  float* f1   = yb   + (size_t)BT * E;        // [BT, FF]
  float* nll  = f1   + (size_t)BT * FF;       // [BT]
  float* logits = (float*)d_out;              // [BT, V] + loss

  embed_kernel<<<BT, 256, 0, stream>>>(idx, wte, wpe, x);

  for (int l = 0; l < Ll; ++l) {
    ln_kernel<<<BT, 256, 0, stream>>>(x, ln1_g + (size_t)l * E, ln1_b + (size_t)l * E, h);
    gemm_wmma<<<dim3(3 * E / 128, BT / 64), 128, 0, stream>>>(
        h, qkv_w + (size_t)l * E * 3 * E, nullptr, nullptr, qkv, BT, 3 * E, E, 0, 0);
    attn_kernel<<<dim3(Tt / 16, Bb * Hh), 32, 0, stream>>>(qkv, yb);
    gemm_wmma<<<dim3(E / 128, BT / 64), 128, 0, stream>>>(
        yb, proj_w + (size_t)l * E * E, proj_b + (size_t)l * E, x, x, BT, E, E, 0, 0);
    ln_kernel<<<BT, 256, 0, stream>>>(x, ln2_g + (size_t)l * E, ln2_b + (size_t)l * E, h);
    gemm_wmma<<<dim3(FF / 128, BT / 64), 128, 0, stream>>>(
        h, fc1_w + (size_t)l * E * FF, fc1_b + (size_t)l * FF, nullptr, f1, BT, FF, E, 0, 1);
    gemm_wmma<<<dim3(E / 128, BT / 64), 128, 0, stream>>>(
        f1, fc2_w + (size_t)l * FF * E, fc2_b + (size_t)l * E, x, x, BT, E, FF, 0, 0);
  }

  ln_kernel<<<BT, 256, 0, stream>>>(x, lnf_g, lnf_b, h);
  // logits = h @ wte^T (tied head), B stored [V, E] -> transB
  gemm_wmma<<<dim3((Vv + 127) / 128, BT / 64), 128, 0, stream>>>(
      h, wte, nullptr, nullptr, logits, BT, Vv, E, 1, 0);

  nll_kernel<<<BT, 256, 0, stream>>>(logits, targets, nll);
  loss_reduce<<<1, 256, 0, stream>>>(nll, logits + (size_t)BT * Vv);
}
